// TextCNNVAttnV2_22582938042450
// MI455X (gfx1250) — compile-verified, matrix operand order checked by reading, not run on hardware
//
#include <hip/hip_runtime.h>
#include <hip/hip_bf16.h>
#include <stdint.h>

// ---------------- CDNA5 WMMA plumbing (wave32, 16x16x32 bf16, f32 accum) ----
typedef __bf16 v16bf __attribute__((ext_vector_type(16)));
typedef float  v8f   __attribute__((ext_vector_type(8)));

union FragU { v16bf v; uint4 q[2]; };

// A/B fragment loader. Both operands are K-major ([rows, K] row-major), so the
// WMMA computes Y = A * W^T with the ISA's native lane striping:
// lane l holds row (l&15); elements 0..7 = K kb..kb+7, 8..15 = K 16+kb..16+kb+7,
// kb = 8*(l>>4).  Two 16-byte loads per fragment.
__device__ __forceinline__ v16bf load_frag(const __bf16* base, int ld) {
  int lane = threadIdx.x & 31;
  const __bf16* p = base + (size_t)(lane & 15) * (size_t)ld + ((lane >> 4) << 3);
  FragU u;
  u.q[0] = *(const uint4*)(p);
  u.q[1] = *(const uint4*)(p + 16);
  return u.v;
}

__device__ __forceinline__ v8f wmma_bf(v16bf a, v16bf b, v8f c) {
  return __builtin_amdgcn_wmma_f32_16x16x32_bf16(false, a, false, b, (short)0, c,
                                                 false, false);
}

#define ZERO8 {0.f,0.f,0.f,0.f,0.f,0.f,0.f,0.f}

// ---------------- problem constants ----------------------------------------
constexpr int Bc  = 8,  Sc = 1024, Cc = 512, Ddim = 512, NCc = 512, Hh = 8;
constexpr int M1  = Bc * Sc;     // 8192 tokens at full resolution
constexpr int S4  = Sc / 4;      // 256
constexpr int M4c = Bc * S4;     // 2048

// ---------------- generic WMMA GEMM: Y[m,n] = alpha*sum_k A[m,k]W[n,k] + bias
// block = 256 threads = 8 waves; wave -> 32x64 tile (2x4 fragments, 8 WMMA per
// 6 fragment loads); block tile 128x128; grid = (N/128, M/128).
// Optional residual add, fp32 and/or bf16 stores.
__global__ __launch_bounds__(256)
void k_gemm(const __bf16* __restrict__ A, const __bf16* __restrict__ W,
            const float* __restrict__ bias, int bias_mod,
            const float* __restrict__ residual, float alpha,
            float* __restrict__ outF, __bf16* __restrict__ outB,
            int M, int N, int K) {
  const int w    = threadIdx.x >> 5;
  const int lane = threadIdx.x & 31;
  const int m0 = blockIdx.y * 128 + (w & 3) * 32;   // 4 wave-rows of 32
  const int n0 = blockIdx.x * 128 + (w >> 2) * 64;  // 2 wave-cols of 64

  const __bf16* Ap = A + (size_t)m0 * K;
  const __bf16* Wp = W + (size_t)n0 * K;

  v8f acc[2][4];
#pragma unroll
  for (int i = 0; i < 2; ++i)
#pragma unroll
    for (int j = 0; j < 4; ++j) acc[i][j] = ZERO8;

#pragma unroll 2
  for (int k = 0; k < K; k += 32) {
    __builtin_prefetch(Ap + k + 256, 0, 3);          // near-scope prefetch
    __builtin_prefetch(Wp + k + 256, 0, 3);
    v16bf af[2], bf[4];
#pragma unroll
    for (int i = 0; i < 2; ++i)
      af[i] = load_frag(Ap + (size_t)(16 * i) * K + k, K);
#pragma unroll
    for (int j = 0; j < 4; ++j)
      bf[j] = load_frag(Wp + (size_t)(16 * j) * K + k, K);
#pragma unroll
    for (int i = 0; i < 2; ++i)
#pragma unroll
      for (int j = 0; j < 4; ++j) acc[i][j] = wmma_bf(af[i], bf[j], acc[i][j]);
  }

  const int col = lane & 15, rb = (lane >> 4) * 8;
#pragma unroll
  for (int i = 0; i < 2; ++i) {
#pragma unroll
    for (int j = 0; j < 4; ++j) {
      const int mr = m0 + 16 * i + rb;
      const int nc = n0 + 16 * j + col;
      // nc < 2*bias_mod for every call site -> conditional subtract, no idiv
      const int bix = (nc < bias_mod) ? nc : nc - bias_mod;
      float bval = bias[bix];
#pragma unroll
      for (int r = 0; r < 8; ++r) {
        size_t o = (size_t)(mr + r) * N + nc;
        float v = acc[i][j][r] * alpha + bval;
        if (residual) v += residual[o];
        if (outF) outF[o] = v;
        if (outB) outB[o] = (__bf16)v;
      }
    }
  }
}

// ---------------- flash-style MHA (one wave per 16-query tile per (b,h)) ----
// qkv: bf16 [B*S, 3C]; out: bf16 [B*S, C]. head_dim = 64, scale = 1/8.
// V tiles are staged global->LDS with the CDNA5 async path (ASYNCcnt).
__global__ __launch_bounds__(32)
void k_attention(const __bf16* __restrict__ qkv, __bf16* __restrict__ out,
                 int Slen) {
  const int ld = 3 * Cc;
  const int b = blockIdx.x / Hh, h = blockIdx.x % Hh;
  const int q0 = blockIdx.y * 16;
  const int lane = threadIdx.x;
  const int col = lane & 15, rb = (lane >> 4) * 8;

  __shared__ __align__(16) __bf16 Pm[16 * 32];   // P tile (q x k) for A-frag
  __shared__ __align__(16) __bf16 Vld[32 * 64];  // V tile (k x d), row-major

  const __bf16* qb = qkv + (size_t)(b * Slen + q0) * ld + h * 64;
  v16bf aq0 = load_frag(qb, ld);
  v16bf aq1 = load_frag(qb + 32, ld);

  v8f acc0 = ZERO8, acc1 = ZERO8, acc2 = ZERO8, acc3 = ZERO8;
  float m8[8], l8[8];
#pragma unroll
  for (int r = 0; r < 8; ++r) { m8[r] = -1e30f; l8[r] = 0.f; }

  const __bf16* kAll = qkv + (size_t)(b * Slen) * ld + Cc + h * 64;
  const __bf16* vAll = qkv + (size_t)(b * Slen) * ld + 2 * Cc + h * 64;
  const unsigned ldsV = (unsigned)(uintptr_t)(void*)Vld;

  for (int kt = 0; kt < Slen; kt += 32) {
    const __bf16* kb = kAll + (size_t)kt * ld;
    v16bf b00 = load_frag(kb, ld);
    v16bf b01 = load_frag(kb + 32, ld);
    v16bf b10 = load_frag(kb + (size_t)16 * ld, ld);
    v16bf b11 = load_frag(kb + (size_t)16 * ld + 32, ld);
    v8f s0 = ZERO8, s1 = ZERO8;
    s0 = wmma_bf(aq0, b00, s0);  s0 = wmma_bf(aq1, b01, s0);
    s1 = wmma_bf(aq0, b10, s1);  s1 = wmma_bf(aq1, b11, s1);

    __syncthreads();   // prior-iteration LDS reads done before we overwrite
    // ---- async-copy V tile [32 keys x 64 dims] (4 KB) global -> LDS ------
    {
      const __bf16* vtile = vAll + (size_t)kt * ld;
#pragma unroll
      for (int j = 0; j < 8; ++j) {
        int chunk = j * 32 + lane;           // 256 x 16B chunks
        int row = chunk >> 3, seg = chunk & 7;
        unsigned laddr = ldsV + row * 128 + seg * 16;
        const __bf16* g = vtile + (size_t)row * ld + seg * 8;
        asm volatile("global_load_async_to_lds_b128 %0, %1, off"
                     :: "v"(laddr), "v"((unsigned long long)(uintptr_t)g)
                     : "memory");
      }
    }

    float p0[8], p1[8];
#pragma unroll
    for (int r = 0; r < 8; ++r) {
      float x0 = s0[r] * 0.125f, x1 = s1[r] * 0.125f;
      float t = fmaxf(x0, x1);
      t = fmaxf(t, __shfl_xor(t, 1, 32));
      t = fmaxf(t, __shfl_xor(t, 2, 32));
      t = fmaxf(t, __shfl_xor(t, 4, 32));
      t = fmaxf(t, __shfl_xor(t, 8, 32));
      float mnew = fmaxf(m8[r], t);
      float al = __expf(m8[r] - mnew);
      m8[r] = mnew;
      p0[r] = __expf(x0 - mnew);
      p1[r] = __expf(x1 - mnew);
      float rs = p0[r] + p1[r];
      rs += __shfl_xor(rs, 1, 32);
      rs += __shfl_xor(rs, 2, 32);
      rs += __shfl_xor(rs, 4, 32);
      rs += __shfl_xor(rs, 8, 32);
      l8[r] = l8[r] * al + rs;
      acc0[r] *= al; acc1[r] *= al; acc2[r] *= al; acc3[r] *= al;
    }

#pragma unroll
    for (int r = 0; r < 8; ++r) {
      Pm[(rb + r) * 32 + col]      = (__bf16)p0[r];
      Pm[(rb + r) * 32 + col + 16] = (__bf16)p1[r];
    }
    asm volatile("s_wait_asynccnt 0" ::: "memory");   // V tile resident in LDS
    __syncthreads();

    v16bf ap = load_frag(Pm, 32);
    // B-fragments of V gathered transposed out of the row-major LDS tile:
    // lane holds n = d, walks K (key rows) with the ISA striping.
#pragma unroll
    for (int j = 0; j < 4; ++j) {
      const int d = j * 16 + col;
      const int kbs = (lane >> 4) * 8;
      v16bf bv;
#pragma unroll
      for (int e = 0; e < 8; ++e) bv[e] = Vld[(kbs + e) * 64 + d];
#pragma unroll
      for (int e = 0; e < 8; ++e) bv[8 + e] = Vld[(16 + kbs + e) * 64 + d];
      if (j == 0) acc0 = wmma_bf(ap, bv, acc0);
      if (j == 1) acc1 = wmma_bf(ap, bv, acc1);
      if (j == 2) acc2 = wmma_bf(ap, bv, acc2);
      if (j == 3) acc3 = wmma_bf(ap, bv, acc3);
    }
  }

#pragma unroll
  for (int r = 0; r < 8; ++r) {
    float inv = 1.f / l8[r];
    size_t o = (size_t)(b * Slen + q0 + rb + r) * Cc + h * 64 + col;
    out[o]      = (__bf16)(acc0[r] * inv);
    out[o + 16] = (__bf16)(acc1[r] * inv);
    out[o + 32] = (__bf16)(acc2[r] * inv);
    out[o + 48] = (__bf16)(acc3[r] * inv);
  }
}

// ---------------- elementwise / reshape helpers -----------------------------
__global__ void k_embed(const int* __restrict__ x, const float* __restrict__ emb,
                        float* __restrict__ out, size_t total) {
  size_t i = (size_t)blockIdx.x * 256 + threadIdx.x;
  if (i >= total) return;
  out[i] = emb[(size_t)x[i / Cc] * Cc + (i % Cc)];
}

__global__ void k_tobf(const float* __restrict__ in, __bf16* __restrict__ out,
                       size_t n, int elu) {
  size_t i = (size_t)blockIdx.x * 256 + threadIdx.x;
  if (i >= n) return;
  float v = in[i];
  if (elu) v = v > 0.f ? v : (__expf(v) - 1.f);
  out[i] = (__bf16)v;
}

// build [B*L, 3C] im2col rows for k=3 pad=1 conv over per-batch segments
__global__ void k_pad3(const __bf16* __restrict__ x, __bf16* __restrict__ out,
                       int L, size_t total) {
  size_t i = (size_t)blockIdx.x * 256 + threadIdx.x;
  if (i >= total) return;
  int c = (int)(i % Cc);
  size_t r = i / Cc;
  int k = (int)(r % 3);
  size_t m = r / 3;
  int t = (int)(m % L);
  size_t b = m / L;
  int ts = t + k - 1;
  out[i] = (ts >= 0 && ts < L) ? x[(b * L + ts) * (size_t)Cc + c] : (__bf16)0.f;
}

// W[o,c,k] (f32) -> bf16 [o, k*Ci + c]  (covers k=1 plain convert too)
__global__ void k_repack_ock(const float* __restrict__ w, __bf16* __restrict__ o,
                             int Ci, int K, size_t total) {
  size_t i = (size_t)blockIdx.x * 256 + threadIdx.x;
  if (i >= total) return;
  size_t oo = i / ((size_t)K * Ci);
  size_t rem = i % ((size_t)K * Ci);
  int k = (int)(rem / Ci), c = (int)(rem % Ci);
  o[i] = (__bf16)w[oo * (size_t)Ci * K + (size_t)c * K + k];
}

// ConvTranspose weight w[c,o,k] -> bf16 [(k*Co+o), Ci]
__global__ void k_repack_up(const float* __restrict__ w, __bf16* __restrict__ o,
                            int Ci, int Co, int K, size_t total) {
  size_t i = (size_t)blockIdx.x * 256 + threadIdx.x;
  if (i >= total) return;
  int c = (int)(i % Ci);
  size_t r = i / Ci;
  int oo = (int)(r % Co);
  int k = (int)(r / Co);
  o[i] = (__bf16)w[((size_t)c * Co + oo) * K + k];
}

__global__ void k_repack_cbT(const float* __restrict__ cb, __bf16* __restrict__ o,
                             size_t total) {
  size_t i = (size_t)blockIdx.x * 256 + threadIdx.x;
  if (i >= total) return;
  int d = (int)(i % Ddim);
  int n = (int)(i / Ddim);
  o[i] = (__bf16)cb[(size_t)d * NCc + n];
}

__global__ void k_cbnorm(const float* __restrict__ cb, float* __restrict__ out) {
  int n = blockIdx.x * 256 + threadIdx.x;
  if (n >= NCc) return;
  float s = 0.f;
  for (int d = 0; d < Ddim; ++d) { float v = cb[(size_t)d * NCc + n]; s += v * v; }
  out[n] = s;
}

__global__ void k_zero(float* p, int n) {
  int i = blockIdx.x * 256 + threadIdx.x;
  if (i < n) p[i] = 0.f;
}

// ---------------- rowwise kernels (C = 512 fixed, block 256) ----------------
__global__ __launch_bounds__(256)
void k_layernorm(const float* __restrict__ x, const float* __restrict__ g,
                 const float* __restrict__ b, float* __restrict__ out) {
  __shared__ float red[256];
  const int row = blockIdx.x, tid = threadIdx.x;
  const float* xr = x + (size_t)row * Cc;
  float a0 = xr[tid], a1 = xr[tid + 256];
  red[tid] = a0 + a1; __syncthreads();
  for (int off = 128; off; off >>= 1) {
    if (tid < off) red[tid] += red[tid + off];
    __syncthreads();
  }
  float mean = red[0] * (1.f / Cc); __syncthreads();
  float d0 = a0 - mean, d1 = a1 - mean;
  red[tid] = d0 * d0 + d1 * d1; __syncthreads();
  for (int off = 128; off; off >>= 1) {
    if (tid < off) red[tid] += red[tid + off];
    __syncthreads();
  }
  float inv = rsqrtf(red[0] * (1.f / Cc) + 1e-5f);
  out[(size_t)row * Cc + tid]       = d0 * inv * g[tid]       + b[tid];
  out[(size_t)row * Cc + tid + 256] = d1 * inv * g[tid + 256] + b[tid + 256];
}

__global__ __launch_bounds__(256)
void k_logsoftmax(const float* __restrict__ x, float* __restrict__ out) {
  __shared__ float red[256];
  const int row = blockIdx.x, tid = threadIdx.x;
  const float* xr = x + (size_t)row * Cc;
  float a0 = xr[tid], a1 = xr[tid + 256];
  red[tid] = fmaxf(a0, a1); __syncthreads();
  for (int off = 128; off; off >>= 1) {
    if (tid < off) red[tid] = fmaxf(red[tid], red[tid + off]);
    __syncthreads();
  }
  float mx = red[0]; __syncthreads();
  red[tid] = __expf(a0 - mx) + __expf(a1 - mx); __syncthreads();
  for (int off = 128; off; off >>= 1) {
    if (tid < off) red[tid] += red[tid + off];
    __syncthreads();
  }
  float ls = __logf(red[0]);
  out[(size_t)row * Cc + tid]       = (a0 - mx) - ls;
  out[(size_t)row * Cc + tid + 256] = (a1 - mx) - ls;
}

__global__ __launch_bounds__(256)
void k_argmin(const float* __restrict__ dist, int* __restrict__ ids) {
  __shared__ float bv[256];
  __shared__ int   bi[256];
  const int row = blockIdx.x, tid = threadIdx.x;
  const float* dr = dist + (size_t)row * NCc;
  float v0 = dr[tid], v1 = dr[tid + 256];
  float best = v0; int idx = tid;
  if (v1 < best) { best = v1; idx = tid + 256; }
  bv[tid] = best; bi[tid] = idx; __syncthreads();
  for (int off = 128; off; off >>= 1) {
    if (tid < off) {
      if (bv[tid + off] < bv[tid] ||
          (bv[tid + off] == bv[tid] && bi[tid + off] < bi[tid])) {
        bv[tid] = bv[tid + off]; bi[tid] = bi[tid + off];
      }
    }
    __syncthreads();
  }
  if (tid == 0) ids[row] = bi[0];
}

__global__ __launch_bounds__(256)
void k_vq(const float* __restrict__ cb, const int* __restrict__ ids,
          const float* __restrict__ ze, float* __restrict__ zOut,
          __bf16* __restrict__ zbf, float* __restrict__ diff,
          float* __restrict__ idsOut) {
  __shared__ float red[256];
  const int m = blockIdx.x, tid = threadIdx.x;
  const int id = ids[m];
  float local = 0.f;
  for (int d = tid; d < Ddim; d += 256) {
    float q = cb[(size_t)d * NCc + id];
    size_t o = (size_t)m * Ddim + d;
    zOut[o] = q;
    zbf[o]  = (__bf16)q;
    float df = q - ze[o];
    local += df * df;
  }
  red[tid] = local; __syncthreads();
  for (int off = 128; off; off >>= 1) {
    if (tid < off) red[tid] += red[tid + off];
    __syncthreads();
  }
  if (tid == 0) {
    atomicAdd(diff, red[0] * (1.f / ((float)M4c * (float)Ddim)));
    idsOut[m] = (float)id;
  }
}

// ---------------- orchestration --------------------------------------------
extern "C" void kernel_launch(void* const* d_in, const int* in_sizes, int n_in,
                              void* d_out, int out_size, void* d_ws, size_t ws_size,
                              hipStream_t stream) {
  (void)in_sizes; (void)n_in; (void)out_size; (void)ws_size;
  // ---- inputs (setup_inputs order) ----
  const int*   x          = (const int*)  d_in[0];
  const float* embed      = (const float*)d_in[1];
  const float* enc_res_w  = (const float*)d_in[2];
  const float* enc_res_b  = (const float*)d_in[3];
  const float* enc_in_w   = (const float*)d_in[4];
  const float* enc_in_b   = (const float*)d_in[5];
  const float* enc_out_w  = (const float*)d_in[6];
  const float* enc_out_b  = (const float*)d_in[7];
  const float* enc_ln_g   = (const float*)d_in[8];
  const float* enc_ln_b   = (const float*)d_in[9];
  const float* enc_down_w = (const float*)d_in[10];
  const float* enc_down_b = (const float*)d_in[11];
  const float* e2q_w      = (const float*)d_in[12];
  const float* e2q_b      = (const float*)d_in[13];
  const float* codebook   = (const float*)d_in[14];
  const float* q2d_w      = (const float*)d_in[15];
  const float* q2d_b      = (const float*)d_in[16];
  const float* dec_res_w  = (const float*)d_in[17];
  const float* dec_res_b  = (const float*)d_in[18];
  const float* dec_in_w   = (const float*)d_in[19];
  const float* dec_in_b   = (const float*)d_in[20];
  const float* dec_out_w  = (const float*)d_in[21];
  const float* dec_out_b  = (const float*)d_in[22];
  const float* dec_ln_g   = (const float*)d_in[23];
  const float* dec_ln_b   = (const float*)d_in[24];
  const float* dec_up_w   = (const float*)d_in[25];
  const float* dec_up_b   = (const float*)d_in[26];

  // ---- outputs: logp | z | diff | ids ----
  float* outLogp = (float*)d_out;
  float* outZ    = outLogp + (size_t)M1 * Cc;
  float* outDiff = outZ + (size_t)M4c * Ddim;
  float* outIds  = outDiff + 1;

  // ---- workspace bump allocator ----
  char* ws = (char*)d_ws;
  size_t off = 0;
  auto alloc = [&](size_t bytes) -> void* {
    void* p = ws + off;
    off = (off + bytes + 255) & ~size_t(255);
    return p;
  };
  float*  F0   = (float*) alloc((size_t)M1 * Cc * 4);
  float*  F1   = (float*) alloc((size_t)M1 * Cc * 4);
  float*  F2   = (float*) alloc((size_t)M1 * Cc * 4);
  __bf16* BFA  = (__bf16*)alloc((size_t)M1 * Cc * 2);      // GEMM A operand
  __bf16* BQKV = (__bf16*)alloc((size_t)M1 * 3 * Cc * 2);  // qkv bf16
  __bf16* BAO  = (__bf16*)alloc((size_t)M1 * Cc * 2);      // attn out / im2col
  float*  DIST = (float*) alloc((size_t)M4c * NCc * 4);
  float*  CBN  = (float*) alloc((size_t)NCc * 4);
  int*    IDS  = (int*)   alloc((size_t)M4c * 4);
  __bf16* Wencres = (__bf16*)alloc((size_t)3 * Cc * Cc * 2);
  __bf16* Wencin  = (__bf16*)alloc((size_t)3 * Cc * Cc * 2);
  __bf16* Wencout = (__bf16*)alloc((size_t)Cc * Cc * 2);
  __bf16* Wdown   = (__bf16*)alloc((size_t)2 * Cc * 2 * Cc * 2);
  __bf16* We2q    = (__bf16*)alloc((size_t)Ddim * 3 * Cc * 2);
  __bf16* WcbT    = (__bf16*)alloc((size_t)NCc * Ddim * 2);
  __bf16* Wq2d    = (__bf16*)alloc((size_t)Cc * 3 * Ddim * 2);
  __bf16* Wdecres = (__bf16*)alloc((size_t)2 * 3 * Cc * Cc * 2);
  __bf16* Wdecin  = (__bf16*)alloc((size_t)2 * 3 * Cc * Cc * 2);
  __bf16* Wdecout = (__bf16*)alloc((size_t)2 * Cc * Cc * 2);
  __bf16* Wdecup  = (__bf16*)alloc((size_t)2 * 2 * Cc * Cc * 2);

  auto grid1 = [](size_t n) { return (unsigned)((n + 255) / 256); };
  auto repack = [&](const float* w, __bf16* o, size_t O, int Ci, int K) {
    size_t tot = O * (size_t)Ci * K;
    k_repack_ock<<<grid1(tot), 256, 0, stream>>>(w, o, Ci, K, tot);
  };
  auto tobf = [&](const float* in, __bf16* o, size_t n, int elu) {
    k_tobf<<<grid1(n), 256, 0, stream>>>(in, o, n, elu);
  };
  auto gemm = [&](const __bf16* A, const __bf16* W, const float* bias,
                  int bias_mod, const float* res, float alpha,
                  float* oF, __bf16* oB, int M, int N, int K) {
    dim3 g(N / 128, M / 128);
    k_gemm<<<g, 256, 0, stream>>>(A, W, bias, bias_mod, res, alpha, oF, oB, M, N, K);
  };

  // ---- weight repacks (bf16, [N,K] K-major) ----
  repack(enc_res_w, Wencres, 3 * Cc, Cc, 1);
  repack(enc_in_w,  Wencin,  3 * Cc, Cc, 1);
  repack(enc_out_w, Wencout, Cc, Cc, 1);
  repack(enc_down_w, Wdown,  2 * Cc, Cc, 2);          // [i*C+o, k*C+c]
  repack(e2q_w, We2q, Ddim, Cc, 3);
  repack(q2d_w, Wq2d, Cc, Ddim, 3);
  repack(dec_res_w, Wdecres, 2 * 3 * Cc, Cc, 1);
  repack(dec_in_w,  Wdecin,  2 * 3 * Cc, Cc, 1);
  repack(dec_out_w, Wdecout, 2 * Cc, Cc, 1);
  for (int i = 0; i < 2; ++i) {
    size_t tot = (size_t)2 * Cc * Cc;
    k_repack_up<<<grid1(tot), 256, 0, stream>>>(
        dec_up_w + (size_t)i * Cc * Cc * 2, Wdecup + (size_t)i * 2 * Cc * Cc,
        Cc, Cc, 2, tot);
  }
  {
    size_t tot = (size_t)NCc * Ddim;
    k_repack_cbT<<<grid1(tot), 256, 0, stream>>>(codebook, WcbT, tot);
  }
  k_cbnorm<<<2, 256, 0, stream>>>(codebook, CBN);
  k_zero<<<1, 256, 0, stream>>>(outDiff, 1);

  const size_t nM1C = (size_t)M1 * Cc;
  const size_t nM4D = (size_t)M4c * Ddim;

  // ================= encoder =================
  k_embed<<<grid1(nM1C), 256, 0, stream>>>(x, embed, F0, nM1C);
  // resblock
  tobf(F0, BFA, nM1C, 1);
  gemm(BFA, Wencres,                 enc_res_b,          Cc, nullptr, 1.f, F1, nullptr, M1, Cc, Cc);
  tobf(F1, BFA, nM1C, 1);
  gemm(BFA, Wencres + (size_t)Cc*Cc, enc_res_b + Cc,     Cc, nullptr, 1.f, F2, nullptr, M1, Cc, Cc);
  tobf(F2, BFA, nM1C, 1);
  gemm(BFA, Wencres + (size_t)2*Cc*Cc, enc_res_b + 2*Cc, Cc, F0,      1.f, F1, nullptr, M1, Cc, Cc);
  // MHA + LN  (X = F1)
  tobf(F1, BFA, nM1C, 0);
  gemm(BFA, Wencin, enc_in_b, 3 * Cc, nullptr, 1.f, nullptr, BQKV, M1, 3 * Cc, Cc);
  k_attention<<<dim3(Bc * Hh, Sc / 16), 32, 0, stream>>>(BQKV, BAO, Sc);
  gemm(BAO, Wencout, enc_out_b, Cc, F1, 1.f, F2, nullptr, M1, Cc, Cc);
  k_layernorm<<<M1, 256, 0, stream>>>(F2, enc_ln_g, enc_ln_b, F0);
  // down-convs: [M,C] viewed as [M/2, 2C]
  tobf(F0, BFA, nM1C, 0);
  gemm(BFA, Wdown, enc_down_b, Cc, nullptr, 1.f, F1, nullptr, M1 / 2, Cc, 2 * Cc);
  tobf(F1, BFA, (size_t)(M1 / 2) * Cc, 0);
  gemm(BFA, Wdown + (size_t)Cc * 2 * Cc, enc_down_b + Cc, Cc, nullptr, 1.f, F2,
       nullptr, M4c, Cc, 2 * Cc);
  // e2q (k=3 pad=1)
  tobf(F2, BFA, nM4D, 0);
  {
    size_t tot = (size_t)M4c * 3 * Cc;
    k_pad3<<<grid1(tot), 256, 0, stream>>>(BFA, BAO, S4, tot);
  }
  gemm(BAO, We2q, e2q_b, Ddim, nullptr, 1.f, F1, nullptr, M4c, Ddim, 3 * Cc);
  // ================= VQ =================
  tobf(F1, BFA, nM4D, 0);
  gemm(BFA, WcbT, CBN, NCc, nullptr, -2.f, DIST, nullptr, M4c, NCc, Ddim);
  k_argmin<<<M4c, 256, 0, stream>>>(DIST, IDS);
  k_vq<<<M4c, 256, 0, stream>>>(codebook, IDS, F1, outZ, BFA, outDiff, outIds);
  // ================= decoder =================
  {
    size_t tot = (size_t)M4c * 3 * Ddim;
    k_pad3<<<grid1(tot), 256, 0, stream>>>(BFA, BAO, S4, tot);
  }
  gemm(BAO, Wq2d, q2d_b, Cc, nullptr, 1.f, F0, nullptr, M4c, Cc, 3 * Ddim);

  float* Hin = F0; float* Ta = F1; float* Tb = F2;
  int Mcur = M4c, Scur = S4;
  for (int i = 0; i < 2; ++i) {
    const __bf16* Wr = Wdecres + (size_t)i * 3 * Cc * Cc;
    const float*  br = dec_res_b + (size_t)i * 3 * Cc;
    size_t nMC = (size_t)Mcur * Cc;
    // resblock
    tobf(Hin, BFA, nMC, 1);
    gemm(BFA, Wr,                   br,          Cc, nullptr, 1.f, Ta, nullptr, Mcur, Cc, Cc);
    tobf(Ta, BFA, nMC, 1);
    gemm(BFA, Wr + (size_t)Cc*Cc,   br + Cc,     Cc, nullptr, 1.f, Tb, nullptr, Mcur, Cc, Cc);
    tobf(Tb, BFA, nMC, 1);
    gemm(BFA, Wr + (size_t)2*Cc*Cc, br + 2*Cc,   Cc, Hin,     1.f, Ta, nullptr, Mcur, Cc, Cc);
    // MHA + LN  (X = Ta)
    tobf(Ta, BFA, nMC, 0);
    gemm(BFA, Wdecin + (size_t)i * 3 * Cc * Cc, dec_in_b + (size_t)i * 3 * Cc,
         3 * Cc, nullptr, 1.f, nullptr, BQKV, Mcur, 3 * Cc, Cc);
    k_attention<<<dim3(Bc * Hh, Scur / 16), 32, 0, stream>>>(BQKV, BAO, Scur);
    gemm(BAO, Wdecout + (size_t)i * Cc * Cc, dec_out_b + (size_t)i * Cc,
         Cc, Ta, 1.f, Tb, nullptr, Mcur, Cc, Cc);
    k_layernorm<<<Mcur, 256, 0, stream>>>(Tb, dec_ln_g + (size_t)i * Cc,
                                          dec_ln_b + (size_t)i * Cc, Hin);
    // conv_transpose up: [Mcur,C]x[2C,C] written contiguously = [2*Mcur, C]
    tobf(Hin, BFA, nMC, 0);
    gemm(BFA, Wdecup + (size_t)i * 2 * Cc * Cc, dec_up_b + (size_t)i * Cc,
         Cc, nullptr, 1.f, Tb, nullptr, Mcur, 2 * Cc, Cc);
    float* t = Hin; Hin = Tb; Tb = Ta; Ta = t;
    Mcur *= 2; Scur *= 2;
  }
  // final log-softmax over C
  k_logsoftmax<<<M1, 256, 0, stream>>>(Hin, outLogp);
}
